// HeadModule_77094662963970
// MI455X (gfx1250) — matrix operand order repaired and checked by reading, not compiled, not run on hardware
//
#include <hip/hip_runtime.h>

typedef __attribute__((ext_vector_type(16))) _Float16 v16h;
typedef __attribute__((ext_vector_type(8)))  _Float16 v8h;
typedef __attribute__((ext_vector_type(8)))  float    v8f;

#define N_PTS 262144
#define RES 256
#define WAVES_PER_BLOCK 4
#define THREADS (WAVES_PER_BLOCK * 32)
#define TILES_PER_WAVE 4
#define PTS_PER_BLOCK (WAVES_PER_BLOCK * 16 * TILES_PER_WAVE)   // 256
#define NUM_BLOCKS (N_PTS / PTS_PER_BLOCK)                      // 1024

// LDS layout (halfs / floats)
#define HWD1 4096     // dW1 frags: 1 ktile * 8 ntiles * 512 halfs
#define HWD2 16384    // dW2 frags: 4 * 8 * 512
#define HWC1 8192     // cW1 frags: 2 * 8 * 512 (K padded 59->64)
#define HWC2 16384    // cW2 frags: 4 * 8 * 512
#define HWD3 2048     // dW3 frags: 4 ktiles * 1 ntile * 512 (col 0 = dW3, rest 0)
#define HWC3 2048     // cW3 frags: 4 * 1 * 512 (cols 0..2 = cW3)
#define HW_TOTAL (HWD1 + HWD2 + HWC1 + HWC2 + HWD3 + HWC3)      // 49152 halfs = 98304 B
#define BIAS_FLOATS 520                                          // 2080 B (16B aligned)
#define STAGE_HALFS (WAVES_PER_BLOCK * 2 * 16 * 128)             // 16384 halfs = 32768 B
#define SMEM_BYTES (HW_TOTAL * 2 + BIAS_FLOATS * 4 + STAGE_HALFS * 2)  // 133152

__device__ __forceinline__ int imin(int a, int b) { return a < b ? a : b; }
__device__ __forceinline__ int imax(int a, int b) { return a > b ? a : b; }

__device__ __forceinline__ v16h cat16(v8h lo, v8h hi) {
  return __builtin_shufflevector(lo, hi, 0,1,2,3,4,5,6,7,8,9,10,11,12,13,14,15);
}

__device__ __forceinline__ v8f wmma_f16(v16h a, v16h b, v8f c) {
  // v_wmma_f32_16x16x32_f16  D = A(16x32 f16) * B(32x16 f16) + C(16x16 f32)
  return __builtin_amdgcn_wmma_f32_16x16x32_f16(false, a, false, b, (short)0, c, false, false);
}

// A fragment from row-major 16x128 f16 staging (ISA 16-bit A 16x32 layout):
// lane: m = lane&15, hi = lane>>4; elems 0..7 -> K = kt*32 + hi*8 + j,
// elems 8..15 -> K = kt*32 + 16 + hi*8 + (j-8)
__device__ __forceinline__ v16h load_a_frag(const _Float16* S, int kt, int lane) {
  const int m = lane & 15, hi = lane >> 4;
  const _Float16* p = S + m * 128 + kt * 32 + hi * 8;
  v8h lo  = *(const v8h*)p;
  v8h hi8 = *(const v8h*)(p + 16);
  return cat16(lo, hi8);
}

// B fragment from pre-swizzled frag store: lane-contiguous 16 halfs per frag index.
__device__ __forceinline__ v16h load_b_frag(const _Float16* F, int fragIdx, int lane) {
  const _Float16* p = F + ((fragIdx << 5) + lane) * 16;
  v8h lo  = *(const v8h*)p;
  v8h hi8 = *(const v8h*)(p + 8);
  return cat16(lo, hi8);
}

// C init: bias[n] broadcast down the column (c[r] = C[r+hi*8][n], n = nt*16 + (lane&15))
__device__ __forceinline__ v8f bias_row(const float* b, int nt, int lane) {
  float bv = b[(nt << 4) + (lane & 15)];
  v8f r = {bv, bv, bv, bv, bv, bv, bv, bv};
  return r;
}

// ReLU + f16 convert + scatter C fragment into row-major 16x128 staging
__device__ __forceinline__ void store_relu(_Float16* S, int nt, int lane, v8f acc) {
  const int n = (nt << 4) + (lane & 15);
  const int hi = lane >> 4;
#pragma unroll
  for (int r = 0; r < 8; ++r) {
    float v = acc[r];
    v = v > 0.f ? v : 0.f;
    S[(r + hi * 8) * 128 + n] = (_Float16)v;
  }
}

// Convert f32 weights (krows x ncols row-major) to f16, pre-swizzled into
// B-fragment order. Fragment (kt,nt), lane, elem j holds
// W[kt*32 + (lane>>4)*8 + j (+8 for j>=8)][nt*16 + (lane&15)], zero outside bounds.
__device__ __forceinline__ void stage_weight_frags(_Float16* dst, const float* W,
                                                   int ktiles, int ntiles,
                                                   int krows, int ncols, int tid) {
  const int positions = ktiles * ntiles * 32;   // (frag, lane) pairs
  for (int pos = tid; pos < positions; pos += THREADS) {
    int lane = pos & 31;
    int f = pos >> 5;
    int kt = f / ntiles, nt = f % ntiles;
    int hi = lane >> 4, n = lane & 15;
    int col = nt * 16 + n;
    int kbase = kt * 32 + hi * 8;
    v8h lo, hih;
#pragma unroll
    for (int j = 0; j < 8; ++j) {
      int k0 = kbase + j, k1 = kbase + 16 + j;
      lo[j]  = (_Float16)((k0 < krows && col < ncols) ? W[k0 * ncols + col] : 0.f);
      hih[j] = (_Float16)((k1 < krows && col < ncols) ? W[k1 * ncols + col] : 0.f);
    }
    _Float16* d = dst + pos * 16;
    *(v8h*)d = lo;
    *(v8h*)(d + 8) = hih;
  }
}

// Bilinear tap set (offsets + edge-masked weights), matching the reference's
// validity semantics: out-of-range taps contribute 0.
struct TapSet { int o00, o10, o01, o11; float w00, w10, w01, w11; };

__device__ __forceinline__ TapSet make_taps(float gx, float gy) {
  float px = (gx + 1.f) * 0.5f * (float)(RES - 1);
  float py = (gy + 1.f) * 0.5f * (float)(RES - 1);
  float x0f = floorf(px), y0f = floorf(py);
  float fx = px - x0f, fy = py - y0f;
  int x0 = (int)x0f, y0 = (int)y0f;
  int x1 = x0 + 1, y1 = y0 + 1;
  float vx0 = ((unsigned)x0 < (unsigned)RES) ? 1.f : 0.f;
  float vx1 = ((unsigned)x1 < (unsigned)RES) ? 1.f : 0.f;
  float vy0 = ((unsigned)y0 < (unsigned)RES) ? 1.f : 0.f;
  float vy1 = ((unsigned)y1 < (unsigned)RES) ? 1.f : 0.f;
  int xc0 = imin(imax(x0, 0), RES - 1), xc1 = imin(imax(x1, 0), RES - 1);
  int yc0 = imin(imax(y0, 0), RES - 1), yc1 = imin(imax(y1, 0), RES - 1);
  TapSet t;
  t.o00 = yc0 * RES + xc0; t.o10 = yc0 * RES + xc1;
  t.o01 = yc1 * RES + xc0; t.o11 = yc1 * RES + xc1;
  t.w00 = (1.f - fx) * (1.f - fy) * vx0 * vy0;
  t.w10 = fx * (1.f - fy) * vx1 * vy0;
  t.w01 = (1.f - fx) * fy * vx0 * vy1;
  t.w11 = fx * fy * vx1 * vy1;
  return t;
}

__device__ __forceinline__ float samp_c(const float* pc, const TapSet& t) {
  return pc[t.o00] * t.w00 + pc[t.o10] * t.w10 + pc[t.o01] * t.w01 + pc[t.o11] * t.w11;
}

extern "C" __global__ void __launch_bounds__(THREADS, 1)
triplane_head_kernel(const float* __restrict__ pts, const float* __restrict__ vds,
                     const float* __restrict__ tp,
                     const float* __restrict__ dW1, const float* __restrict__ dB1,
                     const float* __restrict__ dW2, const float* __restrict__ dB2,
                     const float* __restrict__ dW3, const float* __restrict__ dB3,
                     const float* __restrict__ cW1, const float* __restrict__ cB1,
                     const float* __restrict__ cW2, const float* __restrict__ cB2,
                     const float* __restrict__ cW3, const float* __restrict__ cB3,
                     float* __restrict__ out)
{
  extern __shared__ _Float16 smem[];
  _Float16* Wd1f = smem;
  _Float16* Wd2f = Wd1f + HWD1;
  _Float16* Wc1f = Wd2f + HWD2;
  _Float16* Wc2f = Wc1f + HWC1;
  _Float16* Wd3f = Wc2f + HWC2;
  _Float16* Wc3f = Wd3f + HWD3;
  float*    biasF = (float*)(Wc3f + HWC3);
  _Float16* stage = (_Float16*)(biasF + BIAS_FLOATS);

  const int tid = threadIdx.x;

  // ---- one-time per-block: f32 -> f16 weight staging, pre-swizzled to B-fragment order ----
  stage_weight_frags(Wd1f, dW1, 1, 8, 32, 128, tid);
  stage_weight_frags(Wd2f, dW2, 4, 8, 128, 128, tid);
  stage_weight_frags(Wc1f, cW1, 2, 8, 59, 128, tid);   // rows 59..63 zero-padded
  stage_weight_frags(Wc2f, cW2, 4, 8, 128, 128, tid);
  stage_weight_frags(Wd3f, dW3, 4, 1, 128, 1, tid);    // output proj, col 0 only
  stage_weight_frags(Wc3f, cW3, 4, 1, 128, 3, tid);    // output proj, cols 0..2
  for (int i = tid; i < 128; i += THREADS) {
    biasF[i]       = dB1[i];
    biasF[128 + i] = dB2[i];
    biasF[256 + i] = cB1[i];
    biasF[384 + i] = cB2[i];
  }
  if (tid == 0) {
    biasF[512] = dB3[0];
    biasF[513] = cB3[0]; biasF[514] = cB3[1]; biasF[515] = cB3[2];
  }
  __syncthreads();

  const int wave = tid >> 5;
  const int lane = tid & 31;
  _Float16* S0 = stage + wave * 4096;   // 16x128 f16: MLP input (feat + view embed)
  _Float16* S1 = S0 + 2048;             // 16x128 f16: hidden activations

  const int p  = lane & 15;
  const int hi = lane >> 4;
  const int n15 = lane & 15;            // column index for C/D fragments

  const float* px_base = tp;
  const float* py_base = tp + 32 * RES * RES;
  const float* pz_base = tp + 64 * RES * RES;

  for (int t = 0; t < TILES_PER_WAVE; ++t) {
    const int tilePt = blockIdx.x * PTS_PER_BLOCK + (wave * TILES_PER_WAVE + t) * 16;
    const int gp = tilePt + p;

    // ---- triplane bilinear sampling: each lane does 16 channels of its point ----
    const float u = pts[gp];
    const float v = pts[N_PTS + gp];
    const float w = pts[2 * N_PTS + gp];
    TapSet tx = make_taps(v, w);   // plane_x sampled at (v, w)
    TapSet ty = make_taps(u, w);   // plane_y at (u, w)
    TapSet tz = make_taps(u, v);   // plane_z at (u, v)
#pragma unroll 4
    for (int i = 0; i < 16; ++i) {
      int c = hi * 16 + i;
      float f = samp_c(px_base + c * (RES * RES), tx)
              + samp_c(py_base + c * (RES * RES), ty)
              + samp_c(pz_base + c * (RES * RES), tz);
      S0[p * 128 + c] = (_Float16)f;
    }

    // ---- view-dir embedding -> S0 cols 32..63 (lanes 0..15 handle one point each) ----
    if (hi == 0) {
      float d0 = vds[gp], d1 = vds[N_PTS + gp], d2 = vds[2 * N_PTS + gp];
      _Float16* row = S0 + p * 128;
      row[32] = (_Float16)d0; row[33] = (_Float16)d1; row[34] = (_Float16)d2;
#pragma unroll
      for (int fq = 0; fq < 4; ++fq) {
        float s = (float)(1 << fq);
        float a0 = d0 * s, a1 = d1 * s, a2 = d2 * s;
        row[35 + fq * 3 + 0] = (_Float16)__sinf(a0);
        row[35 + fq * 3 + 1] = (_Float16)__sinf(a1);
        row[35 + fq * 3 + 2] = (_Float16)__sinf(a2);
        row[47 + fq * 3 + 0] = (_Float16)__cosf(a0);
        row[47 + fq * 3 + 1] = (_Float16)__cosf(a1);
        row[47 + fq * 3 + 2] = (_Float16)__cosf(a2);
      }
      row[59] = (_Float16)0.f; row[60] = (_Float16)0.f; row[61] = (_Float16)0.f;
      row[62] = (_Float16)0.f; row[63] = (_Float16)0.f;
    }
    __builtin_amdgcn_wave_barrier();

    // ---- density MLP layer 1: 32 -> 128 ----
    {
      v16h a0 = load_a_frag(S0, 0, lane);
#pragma unroll
      for (int nt = 0; nt < 8; ++nt) {
        v8f acc = bias_row(biasF + 0, nt, lane);
        acc = wmma_f16(a0, load_b_frag(Wd1f, nt, lane), acc);
        store_relu(S1, nt, lane, acc);
      }
    }
    __builtin_amdgcn_wave_barrier();

    // ---- density layer 2: 128 -> 128 ----
    {
      v16h a0 = load_a_frag(S1, 0, lane);
      v16h a1 = load_a_frag(S1, 1, lane);
      v16h a2 = load_a_frag(S1, 2, lane);
      v16h a3 = load_a_frag(S1, 3, lane);
#pragma unroll
      for (int nt = 0; nt < 8; ++nt) {
        v8f acc = bias_row(biasF + 128, nt, lane);
        acc = wmma_f16(a0, load_b_frag(Wd2f, 0 * 8 + nt, lane), acc);
        acc = wmma_f16(a1, load_b_frag(Wd2f, 1 * 8 + nt, lane), acc);
        acc = wmma_f16(a2, load_b_frag(Wd2f, 2 * 8 + nt, lane), acc);
        acc = wmma_f16(a3, load_b_frag(Wd2f, 3 * 8 + nt, lane), acc);
        store_relu(S1, nt, lane, acc);   // overwrite h1 with h2 (A frags in regs)
      }
    }
    __builtin_amdgcn_wave_barrier();

    // ---- density output layer on WMMA: 128 -> 16 (col 0 = density) ----
    {
      v16h a0 = load_a_frag(S1, 0, lane);
      v16h a1 = load_a_frag(S1, 1, lane);
      v16h a2 = load_a_frag(S1, 2, lane);
      v16h a3 = load_a_frag(S1, 3, lane);
      float bv = (n15 == 0) ? biasF[512] : 0.f;
      v8f acc = {bv, bv, bv, bv, bv, bv, bv, bv};
      acc = wmma_f16(a0, load_b_frag(Wd3f, 0, lane), acc);
      acc = wmma_f16(a1, load_b_frag(Wd3f, 1, lane), acc);
      acc = wmma_f16(a2, load_b_frag(Wd3f, 2, lane), acc);
      acc = wmma_f16(a3, load_b_frag(Wd3f, 3, lane), acc);
      if (n15 == 0) {      // lanes 0 and 16 hold column 0 (rows hi*8 .. hi*8+7)
        float4 lo4 = {acc[0], acc[1], acc[2], acc[3]};
        float4 hi4 = {acc[4], acc[5], acc[6], acc[7]};
        *(float4*)(out + tilePt + hi * 8)     = lo4;
        *(float4*)(out + tilePt + hi * 8 + 4) = hi4;
      }
    }
    __builtin_amdgcn_wave_barrier();

    // ---- color layer 1: 64 -> 128 (2 K-chunks from S0) ----
    {
      v16h a0 = load_a_frag(S0, 0, lane);
      v16h a1 = load_a_frag(S0, 1, lane);
#pragma unroll
      for (int nt = 0; nt < 8; ++nt) {
        v8f acc = bias_row(biasF + 256, nt, lane);
        acc = wmma_f16(a0, load_b_frag(Wc1f, 0 * 8 + nt, lane), acc);
        acc = wmma_f16(a1, load_b_frag(Wc1f, 1 * 8 + nt, lane), acc);
        store_relu(S1, nt, lane, acc);
      }
    }
    __builtin_amdgcn_wave_barrier();

    // ---- color layer 2: 128 -> 128 ----
    {
      v16h a0 = load_a_frag(S1, 0, lane);
      v16h a1 = load_a_frag(S1, 1, lane);
      v16h a2 = load_a_frag(S1, 2, lane);
      v16h a3 = load_a_frag(S1, 3, lane);
#pragma unroll
      for (int nt = 0; nt < 8; ++nt) {
        v8f acc = bias_row(biasF + 384, nt, lane);
        acc = wmma_f16(a0, load_b_frag(Wc2f, 0 * 8 + nt, lane), acc);
        acc = wmma_f16(a1, load_b_frag(Wc2f, 1 * 8 + nt, lane), acc);
        acc = wmma_f16(a2, load_b_frag(Wc2f, 2 * 8 + nt, lane), acc);
        acc = wmma_f16(a3, load_b_frag(Wc2f, 3 * 8 + nt, lane), acc);
        store_relu(S1, nt, lane, acc);
      }
    }
    __builtin_amdgcn_wave_barrier();

    // ---- color output layer on WMMA: 128 -> 16 (cols 0..2 = RGB) ----
    {
      v16h a0 = load_a_frag(S1, 0, lane);
      v16h a1 = load_a_frag(S1, 1, lane);
      v16h a2 = load_a_frag(S1, 2, lane);
      v16h a3 = load_a_frag(S1, 3, lane);
      float bv = (n15 < 3) ? biasF[513 + n15] : 0.f;
      v8f acc = {bv, bv, bv, bv, bv, bv, bv, bv};
      acc = wmma_f16(a0, load_b_frag(Wc3f, 0, lane), acc);
      acc = wmma_f16(a1, load_b_frag(Wc3f, 1, lane), acc);
      acc = wmma_f16(a2, load_b_frag(Wc3f, 2, lane), acc);
      acc = wmma_f16(a3, load_b_frag(Wc3f, 3, lane), acc);
      if (n15 < 3) {       // lanes {0,1,2} and {16,17,18} hold RGB columns
        float* dst = out + (1 + n15) * N_PTS + tilePt + hi * 8;
        float4 lo4 = {acc[0], acc[1], acc[2], acc[3]};
        float4 hi4 = {acc[4], acc[5], acc[6], acc[7]};
        *(float4*)dst       = lo4;
        *(float4*)(dst + 4) = hi4;
      }
    }
    __builtin_amdgcn_wave_barrier();
  }
}

extern "C" void kernel_launch(void* const* d_in, const int* in_sizes, int n_in,
                              void* d_out, int out_size, void* d_ws, size_t ws_size,
                              hipStream_t stream) {
  (void)in_sizes; (void)n_in; (void)out_size; (void)d_ws; (void)ws_size;
  const float* pts = (const float*)d_in[0];
  const float* vds = (const float*)d_in[1];
  const float* tp  = (const float*)d_in[2];
  const float* dW1 = (const float*)d_in[3];
  const float* dB1 = (const float*)d_in[4];
  const float* dW2 = (const float*)d_in[5];
  const float* dB2 = (const float*)d_in[6];
  const float* dW3 = (const float*)d_in[7];
  const float* dB3 = (const float*)d_in[8];
  const float* cW1 = (const float*)d_in[9];
  const float* cB1 = (const float*)d_in[10];
  const float* cW2 = (const float*)d_in[11];
  const float* cB2 = (const float*)d_in[12];
  const float* cW3 = (const float*)d_in[13];
  const float* cB3 = (const float*)d_in[14];
  float* out = (float*)d_out;

  (void)hipFuncSetAttribute((const void*)triplane_head_kernel,
                            hipFuncAttributeMaxDynamicSharedMemorySize, SMEM_BYTES);

  dim3 grid(NUM_BLOCKS, 1, 1);    // 1024 blocks x 128 threads (4 waves x 4 tiles)
  dim3 block(THREADS, 1, 1);
  triplane_head_kernel<<<grid, block, SMEM_BYTES, stream>>>(
      pts, vds, tp, dW1, dB1, dW2, dB2, dW3, dB3,
      cW1, cB1, cW2, cB2, cW3, cB3, out);
}
